// SS2D_27943057228195
// MI455X (gfx1250) — matrix operand order repaired and checked by reading, not compiled, not run on hardware
//
#include <hip/hip_runtime.h>

typedef __attribute__((ext_vector_type(16))) _Float16 v16h;
typedef __attribute__((ext_vector_type(8)))  float    v8f;

#define L_SEQ 2304
#define HW    48
#define DM    96     // D_MODEL
#define DI    192    // D_INNER
#define NST   16     // D_STATE
#define RDT   6      // DT_RANK
#define NDBL  48     // padded (R + 2N) = 38 -> 48
#define BPAD  8      // LDS row padding (f16 elems) to stagger banks

// ---------------------------------------------------------------------------
// Multi-tile WMMA strip: D(16 x NT*16, f32) += A(16xK,f16) * B(K x NT*16,f16)
// B is staged in LDS, column-major: Bs[col][K+BPAD] (f16).
// Per-lane element mapping per CDNA5 ISA 7.12.2 (wave32):
//   A 16x32 f16 : lane holds row M=lane&15; VGPR v half c:
//                 v<4 : K = g*8 + 2v + c ; v>=4 : K = 16 + g*8 + 2(v-4) + c
//   B 32x16 f16 : lane holds col N=lane&15; element j -> j<8 : K = g*8 + j ; j>=8 : K = 16 + g*8 + (j-8)
//   C/D 16x16 f32: VGPR r -> row = r + 8*g, col = lane&15       (g = lane>>4)
// One A fragment per K-chunk feeds NT WMMAs (register-level A reuse).
// ---------------------------------------------------------------------------
template <int K, int NT, typename FA>
__device__ __forceinline__ void wmma_strip(FA loadA, const _Float16* Bs,
                                           int mr, int g, v8f* acc) {
#pragma unroll
  for (int k0 = 0; k0 < K; k0 += 32) {
    v16h av;
#pragma unroll
    for (int j = 0; j < 16; ++j) {
      const int v = j >> 1, c = j & 1;
      const int kA = (v < 4) ? (g * 8 + 2 * v + c) : (16 + g * 8 + 2 * (v - 4) + c);
      av[j] = (_Float16)loadA(k0 + kA);
    }
#pragma unroll
    for (int t = 0; t < NT; ++t) {
      v16h bv;
      const _Float16* bp = Bs + (size_t)(t * 16 + mr) * (K + BPAD) + k0 + g * 8;
#pragma unroll
      for (int j = 0; j < 16; ++j) bv[j] = bp[(j < 8) ? j : (j + 8)];
      acc[t] = __builtin_amdgcn_wmma_f32_16x16x32_f16(false, av, false, bv,
                                                      (short)0, acc[t], false, false);
    }
  }
}

// Cooperative fill of the LDS weight panel (f32 -> f16), 8 k-elems per thread
// so global loads merge to b128 and LDS stores to b128.
template <int K, int NCOLS, typename FB>
__device__ __forceinline__ void fill_B(FB loadBglob, _Float16* Bs) {
#pragma unroll 1
  for (int i = threadIdx.x; i < NCOLS * (K / 8); i += (int)blockDim.x) {
    const int col = i / (K / 8);
    const int k0 = (i % (K / 8)) * 8;
    _Float16* dst = Bs + (size_t)col * (K + BPAD) + k0;
#pragma unroll
    for (int t = 0; t < 8; ++t) dst[t] = (_Float16)loadBglob(col, k0 + t);
  }
  __syncthreads();
}

// ---------------------------------------------------------------------------
// 1) Input projection: xz[db][l][e] = sum_c xs[d][b][l][c] * W_in[d][e][c]
//    Block = 8 waves (8 M-tiles) x 64-col panel; grid 8db * 18mg * 6ng = 864.
// ---------------------------------------------------------------------------
__global__ void k_inproj(const float* __restrict__ x,
                         const float* __restrict__ W_in,
                         float* __restrict__ xz) {
  __shared__ _Float16 Bs[64 * (DM + BPAD)];
  const int bi = blockIdx.x;
  const int db = bi / 108;          // 6*18
  const int r = bi % 108;
  const int ng = r % 6;
  const int mg = r / 6;
  const int d = db >> 1, b = db & 1;
  const int nb = ng * 64;

  auto loadBglob = [&](int col, int k) -> float {
    return W_in[((size_t)d * 2 * DI + nb + col) * DM + k];
  };
  fill_B<DM, 64>(loadBglob, Bs);

  const int wv = threadIdx.x >> 5;
  const int lane = threadIdx.x & 31;
  const int g = lane >> 4, mr = lane & 15;
  const int m0 = (mg * 8 + wv) * 16;

  const int l = m0 + mr;                       // this lane's A row (sequence pos)
  const int p = (d & 1) ? (L_SEQ - 1 - l) : l; // direction reversal
  const int h = (d < 2) ? (p / HW) : (p % HW); // transpose for d>=2
  const int w = (d < 2) ? (p % HW) : (p / HW);
  const float* xb = x + ((size_t)(b * DM) * HW + h) * HW + w;
  auto loadA = [&](int k) -> float { return xb[(size_t)k * HW * HW]; };

  v8f acc[4];
#pragma unroll
  for (int t = 0; t < 4; ++t) acc[t] = (v8f){0.f, 0.f, 0.f, 0.f, 0.f, 0.f, 0.f, 0.f};
  wmma_strip<DM, 4>(loadA, Bs, mr, g, acc);

  float* out = xz + (size_t)db * L_SEQ * (2 * DI);
#pragma unroll
  for (int t = 0; t < 4; ++t) {
    const int col = nb + t * 16 + mr;
#pragma unroll
    for (int rr = 0; rr < 8; ++rr)
      out[(size_t)(m0 + rr + 8 * g) * (2 * DI) + col] = acc[t][rr];
  }
}

// ---------------------------------------------------------------------------
// 2) Depthwise causal conv (k=4) + bias + SiLU on the xi half of xz.
// ---------------------------------------------------------------------------
__global__ void k_conv_silu(const float* __restrict__ xz,
                            const float* __restrict__ conv_w,
                            const float* __restrict__ conv_b,
                            float* __restrict__ xi) {
  const int idx = blockIdx.x * blockDim.x + threadIdx.x;
  if (idx >= 8 * L_SEQ * DI) return;
  const int ch = idx % DI;
  const int l  = (idx / DI) % L_SEQ;
  const int db = idx / (DI * L_SEQ);
  const int d  = db >> 1;
  const float* base = xz + (size_t)db * L_SEQ * (2 * DI);
  float acc = conv_b[d * DI + ch];
#pragma unroll
  for (int j = 0; j < 4; ++j) {
    const int ls = l - 3 + j;
    if (ls >= 0) acc += base[(size_t)ls * (2 * DI) + ch] * conv_w[(d * DI + ch) * 4 + j];
  }
  xi[idx] = acc / (1.f + expf(-acc));  // SiLU
}

// ---------------------------------------------------------------------------
// 3) x_dbl[db][l][n] = sum_k xi[db][l][k] * W_x[d][n][k]  (n<38; cols 38..47=0)
//    Block = 8 waves x full 48-col panel; grid 8db * 18mg = 144.
// ---------------------------------------------------------------------------
__global__ void k_xdbl(const float* __restrict__ xi,
                       const float* __restrict__ W_x,
                       float* __restrict__ xdbl) {
  __shared__ _Float16 Bs[NDBL * (DI + BPAD)];
  const int bi = blockIdx.x;
  const int db = bi / 18;
  const int mg = bi % 18;
  const int d = db >> 1;

  auto loadBglob = [&](int col, int k) -> float {
    return (col < 38) ? W_x[((size_t)d * 38 + col) * DI + k] : 0.f;
  };
  fill_B<DI, NDBL>(loadBglob, Bs);

  const int wv = threadIdx.x >> 5;
  const int lane = threadIdx.x & 31;
  const int g = lane >> 4, mr = lane & 15;
  const int m0 = (mg * 8 + wv) * 16;

  const float* Ab = xi + ((size_t)db * L_SEQ + m0 + mr) * DI;
  auto loadA = [&](int k) -> float { return Ab[k]; };

  v8f acc[3];
#pragma unroll
  for (int t = 0; t < 3; ++t) acc[t] = (v8f){0.f, 0.f, 0.f, 0.f, 0.f, 0.f, 0.f, 0.f};
  wmma_strip<DI, 3>(loadA, Bs, mr, g, acc);

  float* out = xdbl + (size_t)db * L_SEQ * NDBL;
#pragma unroll
  for (int t = 0; t < 3; ++t) {
    const int col = t * 16 + mr;
#pragma unroll
    for (int rr = 0; rr < 8; ++rr)
      out[(size_t)(m0 + rr + 8 * g) * NDBL + col] = acc[t][rr];
  }
}

// ---------------------------------------------------------------------------
// 4) Fused dt-projection + softplus + sequential selective scan + gate.
//    One thread per (db, channel); h[16] kept in registers; L=2304 steps.
// ---------------------------------------------------------------------------
__global__ void k_scan(const float* __restrict__ xz,
                       const float* __restrict__ xi,
                       const float* __restrict__ xdbl,
                       const float* __restrict__ W_dt,
                       const float* __restrict__ b_dt,
                       const float* __restrict__ A_log,
                       const float* __restrict__ D_skip,
                       float* __restrict__ y) {
  const int db = blockIdx.x;        // 0..7
  const int d  = db >> 1;
  const int ch = threadIdx.x;       // 0..191

  float wdt[RDT];
#pragma unroll
  for (int r = 0; r < RDT; ++r) wdt[r] = W_dt[(d * DI + ch) * RDT + r];
  const float bdt = b_dt[d * DI + ch];
  float negA[NST];
#pragma unroll
  for (int n = 0; n < NST; ++n) negA[n] = -expf(A_log[(d * DI + ch) * NST + n]);
  const float dsk = D_skip[d * DI + ch];

  float hst[NST];
#pragma unroll
  for (int n = 0; n < NST; ++n) hst[n] = 0.f;

  const float* xd  = xdbl + (size_t)db * L_SEQ * NDBL;
  const float* xib = xi   + (size_t)db * L_SEQ * DI;
  const float* zb  = xz   + (size_t)db * L_SEQ * (2 * DI) + DI;  // z half
  float* yb        = y    + (size_t)db * L_SEQ * DI;

  for (int l = 0; l < L_SEQ; ++l) {
    const float* row = xd + (size_t)l * NDBL;
    float dtin = bdt;
#pragma unroll
    for (int r = 0; r < RDT; ++r) dtin += row[r] * wdt[r];
    const float dt = (dtin > 20.f) ? dtin : log1pf(expf(dtin));  // softplus
    const float xiv = xib[(size_t)l * DI + ch];
    const float du = dt * xiv;
    float ya = 0.f;
#pragma unroll
    for (int n = 0; n < NST; ++n) {
      const float a = expf(dt * negA[n]);
      hst[n] = a * hst[n] + du * row[RDT + n];       // Bs
      ya += hst[n] * row[RDT + NST + n];             // Cs
    }
    const float zv = zb[(size_t)l * (2 * DI) + ch];
    yb[(size_t)l * DI + ch] = (ya + xiv * dsk) * (zv / (1.f + expf(-zv)));
  }
}

// ---------------------------------------------------------------------------
// 5) Per-direction output projection: ydir[db][l][n] = sum_k y[db][l][k]*W_out[d][n][k]
//    Block = 8 waves x full 96-col panel; grid 8db * 18mg = 144.
// ---------------------------------------------------------------------------
__global__ void k_outproj(const float* __restrict__ y,
                          const float* __restrict__ W_out,
                          float* __restrict__ ydir) {
  __shared__ _Float16 Bs[DM * (DI + BPAD)];
  const int bi = blockIdx.x;
  const int db = bi / 18;
  const int mg = bi % 18;
  const int d = db >> 1;

  auto loadBglob = [&](int col, int k) -> float {
    return W_out[((size_t)d * DM + col) * DI + k];
  };
  fill_B<DI, DM>(loadBglob, Bs);

  const int wv = threadIdx.x >> 5;
  const int lane = threadIdx.x & 31;
  const int g = lane >> 4, mr = lane & 15;
  const int m0 = (mg * 8 + wv) * 16;

  const float* Ab = y + ((size_t)db * L_SEQ + m0 + mr) * DI;
  auto loadA = [&](int k) -> float { return Ab[k]; };

  v8f acc[6];
#pragma unroll
  for (int t = 0; t < 6; ++t) acc[t] = (v8f){0.f, 0.f, 0.f, 0.f, 0.f, 0.f, 0.f, 0.f};
  wmma_strip<DI, 6>(loadA, Bs, mr, g, acc);

  float* out = ydir + (size_t)db * L_SEQ * DM;
#pragma unroll
  for (int t = 0; t < 6; ++t) {
    const int col = t * 16 + mr;
#pragma unroll
    for (int rr = 0; rr < 8; ++rr)
      out[(size_t)(m0 + rr + 8 * g) * DM + col] = acc[t][rr];
  }
}

// ---------------------------------------------------------------------------
// 6) Final projection over concatenated directions (K = 4*96 = 384) with the
//    direction de-permutation folded into the A-gather; output NCHW + bias.
//    Block = 8 waves x full 96-col W_proj panel (73.5KB LDS); grid 36.
// ---------------------------------------------------------------------------
__global__ void k_proj(const float* __restrict__ ydir,
                       const float* __restrict__ W_proj,
                       const float* __restrict__ b_proj,
                       float* __restrict__ out) {
  __shared__ _Float16 Bs[DM * (4 * DM + BPAD)];
  const int mg = blockIdx.x;        // 0..35

  auto loadBglob = [&](int col, int k) -> float {
    return W_proj[(size_t)col * (4 * DM) + k];
  };
  fill_B<4 * DM, DM>(loadBglob, Bs);

  const int wv = threadIdx.x >> 5;
  const int lane = threadIdx.x & 31;
  const int g = lane >> 4, mr = lane & 15;
  const int m0 = (mg * 8 + wv) * 16;

  const int m = m0 + mr;            // this lane's A row: pixel index
  const int b = m / L_SEQ;
  const int pix = m % L_SEQ;
  const int h = pix / HW, w = pix % HW;
  const int l_fwd = h * HW + w;
  const int l_t   = w * HW + h;

  auto loadA = [&](int k) -> float {
    const int dir = k / DM, c = k % DM;
    int l;
    if (dir == 0)      l = l_fwd;
    else if (dir == 1) l = L_SEQ - 1 - l_fwd;
    else if (dir == 2) l = l_t;
    else               l = L_SEQ - 1 - l_t;
    return ydir[(((size_t)dir * 2 + b) * L_SEQ + l) * DM + c];
  };

  v8f acc[6];
#pragma unroll
  for (int t = 0; t < 6; ++t) acc[t] = (v8f){0.f, 0.f, 0.f, 0.f, 0.f, 0.f, 0.f, 0.f};
  wmma_strip<4 * DM, 6>(loadA, Bs, mr, g, acc);

#pragma unroll
  for (int t = 0; t < 6; ++t) {
    const int col = t * 16 + mr;
    const float bp = b_proj[col];
#pragma unroll
    for (int rr = 0; rr < 8; ++rr) {
      const int mm = m0 + rr + 8 * g;
      const int bb = mm / L_SEQ;
      const int pp = mm % L_SEQ;
      const int hh = pp / HW, ww = pp % HW;
      out[(((size_t)bb * DM + col) * HW + hh) * HW + ww] = acc[t][rr] + bp;
    }
  }
}

// ---------------------------------------------------------------------------
extern "C" void kernel_launch(void* const* d_in, const int* in_sizes, int n_in,
                              void* d_out, int out_size, void* d_ws, size_t ws_size,
                              hipStream_t stream) {
  (void)in_sizes; (void)n_in; (void)out_size; (void)ws_size;
  const float* x      = (const float*)d_in[0];
  const float* W_in   = (const float*)d_in[1];
  const float* conv_w = (const float*)d_in[2];
  const float* conv_b = (const float*)d_in[3];
  const float* W_x    = (const float*)d_in[4];
  const float* W_dt   = (const float*)d_in[5];
  const float* b_dt   = (const float*)d_in[6];
  const float* A_log  = (const float*)d_in[7];
  const float* D_skip = (const float*)d_in[8];
  const float* W_out  = (const float*)d_in[9];
  const float* W_proj = (const float*)d_in[10];
  const float* b_proj = (const float*)d_in[11];
  float* out = (float*)d_out;

  float* ws   = (float*)d_ws;
  float* xz   = ws;                                   // 8*2304*384 f32
  float* xi   = xz   + (size_t)8 * L_SEQ * (2 * DI);  // 8*2304*192
  float* xdbl = xi   + (size_t)8 * L_SEQ * DI;        // 8*2304*48
  float* y    = xdbl + (size_t)8 * L_SEQ * NDBL;      // 8*2304*192
  float* ydir = y    + (size_t)8 * L_SEQ * DI;        // 8*2304*96

  hipLaunchKernelGGL(k_inproj,    dim3(864),   dim3(256), 0, stream, x, W_in, xz);
  hipLaunchKernelGGL(k_conv_silu, dim3(13824), dim3(256), 0, stream, xz, conv_w, conv_b, xi);
  hipLaunchKernelGGL(k_xdbl,      dim3(144),   dim3(256), 0, stream, xi, W_x, xdbl);
  hipLaunchKernelGGL(k_scan,      dim3(8),     dim3(192), 0, stream,
                     xz, xi, xdbl, W_dt, b_dt, A_log, D_skip, y);
  hipLaunchKernelGGL(k_outproj,   dim3(144),   dim3(256), 0, stream, y, W_out, ydir);
  hipLaunchKernelGGL(k_proj,      dim3(36),    dim3(256), 0, stream, ydir, W_proj, b_proj, out);
}
